// PointNet2MSGCls_64381559767618
// MI455X (gfx1250) — matrix operand order repaired and checked by reading, not compiled, not run on hardware
//
#include <hip/hip_runtime.h>
#include <hip/hip_bf16.h>
#include <cstdint>

#define EPSV 1e-5f

typedef __attribute__((ext_vector_type(16))) _Float16 v16h;
typedef __attribute__((ext_vector_type(8)))  _Float16 v8h;
typedef __attribute__((ext_vector_type(8)))  float    v8f;

struct LayerP {
  const float* w; const float* b; const float* g; const float* bt;
  const float* mn; const float* vr; int cout;
};

// prepped layer: f16 weights (cout x Kp, zero-padded), fused BN scale/shift
struct HLayer {
  const _Float16* wh; const float* sc; const float* sh; int cout; int Kp;
};

__host__ __device__ inline int a32i(int v) { return (v + 31) & ~31; }

// ---------------- transpose (B,3,N) -> (B,N,3) ----------------
__global__ void k_transpose(const float* __restrict__ x, float* __restrict__ xyz, int B, int N) {
  int t = blockIdx.x * blockDim.x + threadIdx.x;
  int tot = B * N * 3;
  if (t >= tot) return;
  int b = t / (N * 3); int r = t - b * N * 3; int n = r / 3; int c = r - n * 3;
  xyz[t] = x[(size_t)b * 3 * N + (size_t)c * N + n];
}

// ---------------- weight f32->f16 (K-padded) + BN fold ----------------
__global__ void k_prep(const float* __restrict__ w, const float* __restrict__ b,
                       const float* __restrict__ g, const float* __restrict__ bt,
                       const float* __restrict__ mn, const float* __restrict__ vr,
                       int cout, int CR, int Kp,
                       _Float16* __restrict__ wh, float* __restrict__ sc,
                       float* __restrict__ sh) {
  int t = blockIdx.x * blockDim.x + threadIdx.x;
  if (t < cout) {
    float s = g[t] * rsqrtf(vr[t] + EPSV);
    sc[t] = s;
    sh[t] = (b[t] - mn[t]) * s + bt[t];
  }
  if (t >= cout * Kp) return;
  int o = t / Kp, k = t - o * Kp;
  wh[t] = (_Float16)((k < CR) ? w[(size_t)o * CR + k] : 0.f);
}

// ---------------- farthest point sampling ----------------
__global__ void k_fps(const float* __restrict__ xyz, int N, int npoint, int* __restrict__ fidx) {
  int b = blockIdx.x;
  const float* p = xyz + (size_t)b * N * 3;
  __shared__ float rmax[256];
  __shared__ int   rarg[256];
  __shared__ int   s_far;
  float dist[8];
  int cnt = 0;
  for (int j = threadIdx.x; j < N; j += 256) { dist[cnt++] = 1e10f; }
  int far = 0;
  for (int it = 0; it < npoint; ++it) {
    if (threadIdx.x == 0) fidx[b * npoint + it] = far;
    float cx = p[far * 3 + 0], cy = p[far * 3 + 1], cz = p[far * 3 + 2];
    float bm = -1.f; int ba = 0; cnt = 0;
    for (int j = threadIdx.x; j < N; j += 256) {
      float dx = p[j * 3 + 0] - cx, dy = p[j * 3 + 1] - cy, dz = p[j * 3 + 2] - cz;
      float d = dx * dx + dy * dy + dz * dz;
      float nd = fminf(dist[cnt], d);
      dist[cnt] = nd; cnt++;
      if (nd > bm) { bm = nd; ba = j; }
    }
    rmax[threadIdx.x] = bm; rarg[threadIdx.x] = ba;
    __syncthreads();
    for (int s = 128; s > 0; s >>= 1) {
      if (threadIdx.x < s) {
        if (rmax[threadIdx.x + s] > rmax[threadIdx.x]) {
          rmax[threadIdx.x] = rmax[threadIdx.x + s];
          rarg[threadIdx.x] = rarg[threadIdx.x + s];
        }
      }
      __syncthreads();
    }
    if (threadIdx.x == 0) s_far = rarg[0];
    __syncthreads();
    far = s_far;
    __syncthreads();
  }
}

// ---------------- gather sampled centers ----------------
__global__ void k_gather(const float* __restrict__ xyz, const int* __restrict__ fidx,
                         int N, int S, int total, float* __restrict__ out) {
  int t = blockIdx.x * blockDim.x + threadIdx.x;
  if (t >= total) return;
  int b = t / S;
  int j = fidx[t];
  out[(size_t)t * 3 + 0] = xyz[((size_t)b * N + j) * 3 + 0];
  out[(size_t)t * 3 + 1] = xyz[((size_t)b * N + j) * 3 + 1];
  out[(size_t)t * 3 + 2] = xyz[((size_t)b * N + j) * 3 + 2];
}

// ---------------- ball query: one wave per (b,s) ----------------
__global__ void k_ballquery(const float* __restrict__ xyz, const float* __restrict__ new_xyz,
                            int N, int S, int K, float r2, int* __restrict__ idx) {
  int bs = blockIdx.x; int b = bs / S;
  int lane = threadIdx.x;
  const float* p = xyz + (size_t)b * N * 3;
  float cx = new_xyz[(size_t)bs * 3 + 0];
  float cy = new_xyz[(size_t)bs * 3 + 1];
  float cz = new_xyz[(size_t)bs * 3 + 2];
  int* out = idx + (size_t)bs * K;
  int count = 0; int first = N;
  for (int base = 0; base < N && count < K; base += 32) {
    int j = base + lane;
    bool inb = false;
    if (j < N) {
      float dx = p[j * 3 + 0] - cx, dy = p[j * 3 + 1] - cy, dz = p[j * 3 + 2] - cz;
      inb = (dx * dx + dy * dy + dz * dz) <= r2;
    }
    unsigned m = (unsigned)__ballot(inb);
    int pos = count + __popc(m & ((1u << lane) - 1u));
    if (inb && pos < K) out[pos] = j;
    if (first == N && m != 0u) first = base + __ffs(m) - 1;
    count += __popc(m);
  }
  if (count > K) count = K;
  if (first == N) first = 0;
  for (int q = count + lane; q < K; q += 32) out[q] = first;
}

// ---------------- wave-level WMMA GEMM ----------------
// A: LDS f16 (M x lds), B: global pre-padded f16 weights (cout x Kp)
__device__ void wave_gemm(const _Float16* __restrict__ src, int lds, int Kp,
                          const HLayer& H, int M,
                          _Float16* __restrict__ dst, int ldd, float* __restrict__ pooled) {
  int lane = threadIdx.x & 31;
  int wv   = threadIdx.x >> 5;
  int nw   = blockDim.x >> 5;
  int hf   = lane >> 4;
  int lm   = lane & 15;
  int mt = M >> 4, nt = H.cout >> 4;
  for (int t = wv; t < mt * nt; t += nw) {
    int m0 = (t % mt) << 4;
    int n  = ((t / mt) << 4) + lm;
    v8f acc = {};
    const _Float16* ap = src + (size_t)(m0 + lm) * lds + 16 * hf;
    const _Float16* wp = H.wh + (size_t)n * Kp + 16 * hf;
    for (int k0 = 0; k0 < Kp; k0 += 32) {
      v16h av = *(const v16h*)(ap + k0);       // K = k0 + 16*hf + [0..16)
      v16h bv = *(const v16h*)(wp + k0);       // K = k0 + 16*hf + [0..16)
      acc = __builtin_amdgcn_wmma_f32_16x16x32_f16(false, av, false, bv,
                                                   (short)0, acc, false, false);
    }
    float scale = H.sc[n];
    float shift = H.sh[n];
    if (dst) {
#pragma unroll
      for (int r = 0; r < 8; ++r) {
        float v = fmaxf(acc[r] * scale + shift, 0.f);
        dst[(size_t)(m0 + r + 8 * hf) * ldd + n] = (_Float16)v;
      }
    } else {
      float lmax = 0.f;
#pragma unroll
      for (int r = 0; r < 8; ++r)
        lmax = fmaxf(lmax, acc[r] * scale + shift);
      lmax = fmaxf(lmax, 0.f);
      atomicMax((int*)&pooled[n], __float_as_int(lmax));  // valid: values >= 0
    }
  }
}

// ---------------- fused group-MLP: gather -> 3x(GEMM+BN+ReLU) -> max-pool ----------------
// mode 0: features = [points[idx] (Cp), xyz[idx]-center (3)]
// mode 1: features = [xyz (3), points (Cp)], direct rows (global SA stage)
__global__ void __launch_bounds__(128) k_sa_fused(
    const float* __restrict__ xyz, const float* __restrict__ pts,
    const float* __restrict__ new_xyz, const int* __restrict__ idx,
    int S, int N, int K, int Cp, int mode,
    HLayer H0, HLayer H1, HLayer H2,
    float* __restrict__ outp, int ostride, int coff) {
  extern __shared__ char smem[];
  int bs = blockIdx.x;
  int b  = bs / S;
  int Cin  = Cp + 3;
  int C0p  = H0.Kp;                  // a32(Cin)
  int C1p  = H1.Kp;                  // a32(H0.cout)
  int C2p  = H2.Kp;                  // a32(H1.cout)
  int padA = C0p > C2p ? C0p : C2p;
  _Float16* bufA = (_Float16*)smem;
  _Float16* bufB = bufA + (size_t)K * padA;
  float* pooled  = (float*)(bufB + (size_t)K * C1p);

  for (int i = threadIdx.x; i < K * padA; i += blockDim.x) bufA[i] = (_Float16)0.f;
  for (int i = threadIdx.x; i < K * C1p;  i += blockDim.x) bufB[i] = (_Float16)0.f;
  __syncthreads();

  float cx = 0.f, cy = 0.f, cz = 0.f;
  if (mode == 0) {
    cx = new_xyz[(size_t)bs * 3 + 0];
    cy = new_xyz[(size_t)bs * 3 + 1];
    cz = new_xyz[(size_t)bs * 3 + 2];
  }
  for (int i = threadIdx.x; i < K * Cin; i += blockDim.x) {
    int k = i / Cin, c = i - k * Cin;
    float v;
    if (mode == 0) {
      int src = idx[(size_t)bs * K + k];
      if (c < Cp) {
        v = pts[((size_t)b * N + src) * Cp + c];
      } else {
        int d = c - Cp;
        float cc = (d == 0) ? cx : (d == 1) ? cy : cz;
        v = xyz[((size_t)b * N + src) * 3 + d] - cc;
      }
    } else {
      if (c < 3) v = xyz[((size_t)b * N + k) * 3 + c];
      else       v = pts[((size_t)b * N + k) * Cp + (c - 3)];
    }
    bufA[(size_t)k * padA + c] = (_Float16)v;
  }
  __syncthreads();

  wave_gemm(bufA, padA, C0p, H0, K, bufB, C1p, nullptr);
  __syncthreads();
  for (int i = threadIdx.x; i < K * C2p; i += blockDim.x) bufA[i] = (_Float16)0.f;
  __syncthreads();
  wave_gemm(bufB, C1p, C1p, H1, K, bufA, C2p, nullptr);
  __syncthreads();
  for (int i = threadIdx.x; i < H2.cout; i += blockDim.x) pooled[i] = 0.f;
  __syncthreads();
  wave_gemm(bufA, C2p, C2p, H2, K, nullptr, 0, pooled);
  __syncthreads();
  for (int c = threadIdx.x; c < H2.cout; c += blockDim.x)
    outp[(size_t)bs * ostride + coff + c] = pooled[c];
}

// ---------------- head: tiny linear+BN+ReLU ----------------
__global__ void k_linear_bn_relu(const float* __restrict__ in, LayerP L, int M, int Cin,
                                 float* __restrict__ out) {
  int t = blockIdx.x * blockDim.x + threadIdx.x;
  if (t >= M * L.cout) return;
  int m = t / L.cout, o = t - m * L.cout;
  const float* wp = L.w + (size_t)o * Cin;
  const float* ip = in + (size_t)m * Cin;
  float acc = 0.f;
  for (int c = 0; c < Cin; ++c) acc += ip[c] * wp[c];
  float scale = L.g[o] * rsqrtf(L.vr[o] + EPSV);
  float v = (acc + L.b[o] - L.mn[o]) * scale + L.bt[o];
  out[t] = fmaxf(v, 0.f);
}

// ---------------- final linear + log_softmax ----------------
__global__ void k_head(const float* __restrict__ in, const float* __restrict__ w,
                       const float* __restrict__ bias, float* __restrict__ out) {
  int b = blockIdx.x;
  __shared__ float lg[40];
  __shared__ float s_lse;
  for (int o = threadIdx.x; o < 40; o += blockDim.x) {
    float acc = bias[o];
    const float* wp = w + (size_t)o * 256;
    const float* ip = in + (size_t)b * 256;
    for (int c = 0; c < 256; ++c) acc += ip[c] * wp[c];
    lg[o] = acc;
  }
  __syncthreads();
  if (threadIdx.x == 0) {
    float mx = lg[0];
    for (int i = 1; i < 40; ++i) mx = fmaxf(mx, lg[i]);
    float sm = 0.f;
    for (int i = 0; i < 40; ++i) sm += expf(lg[i] - mx);
    s_lse = mx + logf(sm);
  }
  __syncthreads();
  for (int o = threadIdx.x; o < 40; o += blockDim.x)
    out[b * 40 + o] = lg[o] - s_lse;
}

__global__ void k_copy(const float* __restrict__ src, float* __restrict__ dst, int n) {
  int t = blockIdx.x * blockDim.x + threadIdx.x;
  if (t < n) dst[t] = src[t];
}

extern "C" void kernel_launch(void* const* d_in, const int* in_sizes, int n_in,
                              void* d_out, int out_size, void* d_ws, size_t ws_size,
                              hipStream_t stream) {
  (void)in_sizes; (void)n_in; (void)out_size; (void)ws_size;
  const int B = 8, N = 2048;
  auto F = [&](int i) { return (const float*)d_in[i]; };
  auto mkL = [&](int b0, int cout) {
    LayerP L; L.b = F(b0); L.bt = F(b0 + 1); L.g = F(b0 + 2);
    L.mn = F(b0 + 3); L.vr = F(b0 + 4); L.w = F(b0 + 5); L.cout = cout; return L;
  };
  // Flattened input order (alphabetical dict keys):
  // 0: x ; lin1(1..6) ; lin2(7..12) ; lin3 b=13 w=14 ; sa(15..32) ; sm1(33..86) ; sm2(87..140)
  const float* x = F(0);
  LayerP lin1 = mkL(1, 512);
  LayerP lin2 = mkL(7, 256);
  const float* lin3_b = F(13);
  const float* lin3_w = F(14);
  LayerP sa0 = mkL(15, 256), sa1 = mkL(21, 512), sa2 = mkL(27, 1024);
  int sm1c[3][3] = {{32, 32, 64}, {64, 64, 128}, {64, 96, 128}};
  int sm2c[3][3] = {{64, 64, 128}, {128, 128, 256}, {128, 128, 256}};
  LayerP sm1[3][3], sm2[3][3];
  for (int i = 0; i < 3; ++i)
    for (int j = 0; j < 3; ++j) {
      sm1[i][j] = mkL(33 + (i * 3 + j) * 6, sm1c[i][j]);
      sm2[i][j] = mkL(87 + (i * 3 + j) * 6, sm2c[i][j]);
    }

  float* wsf = (float*)d_ws;
  size_t o = 0;
  float* xyz  = wsf + o; o += (size_t)B * N * 3;
  float* l1x  = wsf + o; o += (size_t)B * 512 * 3;
  float* l2x  = wsf + o; o += (size_t)B * 128 * 3;
  float* l1p  = wsf + o; o += (size_t)B * 512 * 320;
  float* l2p  = wsf + o; o += (size_t)B * 128 * 640;
  float* feat = wsf + o; o += (size_t)B * 1024;
  float* h1   = wsf + o; o += (size_t)B * 512;
  float* h2   = wsf + o; o += (size_t)B * 256;
  int* fidx1  = (int*)(wsf + o); o += (size_t)B * 512;
  int* fidx2  = (int*)(wsf + o); o += (size_t)B * 128;
  int* idxb   = (int*)(wsf + o); o += (size_t)B * 512 * 128;

  // ----- prep all WMMA layers: f16 weights + folded BN -----
  struct Spec { LayerP L; int CR; };
  Spec specs[15];
  int si = 0;
  for (int br = 0; br < 3; ++br)
    for (int j = 0; j < 3; ++j)
      specs[si++] = { sm1[br][j], (j == 0) ? 3 : sm1c[br][j - 1] };
  for (int br = 0; br < 3; ++br)
    for (int j = 0; j < 3; ++j)
      specs[si++] = { sm2[br][j], (j == 0) ? 323 : sm2c[br][j - 1] };
  specs[si++] = { sa0, 643 };
  specs[si++] = { sa1, 256 };
  specs[si++] = { sa2, 512 };

  size_t halves = 0, flts = 0;
  for (int i = 0; i < 15; ++i) {
    halves += (size_t)specs[i].L.cout * a32i(specs[i].CR);
    flts   += (size_t)2 * specs[i].L.cout;
  }
  _Float16* hwbase = (_Float16*)(wsf + o);
  o += (halves + 1) / 2;               // halves -> float slots
  float* scbase = wsf + o; o += flts;

  HLayer H[15];
  {
    _Float16* hw = hwbase;
    float* sp = scbase;
    for (int i = 0; i < 15; ++i) {
      int Kp = a32i(specs[i].CR);
      int cout = specs[i].L.cout;
      H[i].wh = hw; H[i].sc = sp; H[i].sh = sp + cout;
      H[i].cout = cout; H[i].Kp = Kp;
      int tot = cout * Kp;
      k_prep<<<(tot + 255) / 256, 256, 0, stream>>>(
          specs[i].L.w, specs[i].L.b, specs[i].L.g, specs[i].L.bt,
          specs[i].L.mn, specs[i].L.vr, cout, specs[i].CR, Kp,
          hw, sp, sp + cout);
      hw += (size_t)tot;
      sp += (size_t)2 * cout;
    }
  }
  HLayer (*Hsm1)[3] = (HLayer(*)[3])&H[0];
  HLayer (*Hsm2)[3] = (HLayer(*)[3])&H[9];
  HLayer* Hsa = &H[18 - 18 + 12];      // H[12..14]

  auto smemB = [&](int K, const HLayer& H0h, const HLayer& H1h, const HLayer& H2h) -> size_t {
    int padA = H0h.Kp > H2h.Kp ? H0h.Kp : H2h.Kp;
    return (size_t)2 * K * padA + (size_t)2 * K * H1h.Kp + (size_t)4 * H2h.cout;
  };

  { int tot = B * N * 3;
    k_transpose<<<(tot + 255) / 256, 256, 0, stream>>>(x, xyz, B, N); }

  // ----- level 1 (npoint=512) -----
  k_fps<<<B, 256, 0, stream>>>(xyz, N, 512, fidx1);
  { int tot = B * 512;
    k_gather<<<(tot + 255) / 256, 256, 0, stream>>>(xyz, fidx1, N, 512, tot, l1x); }
  {
    float r1[3] = {0.1f, 0.2f, 0.4f};
    int   k1[3] = {16, 32, 128};
    int   c1off[3] = {0, 64, 192};
    for (int br = 0; br < 3; ++br) {
      k_ballquery<<<B * 512, 32, 0, stream>>>(xyz, l1x, N, 512, k1[br], r1[br] * r1[br], idxb);
      size_t sm = smemB(k1[br], Hsm1[br][0], Hsm1[br][1], Hsm1[br][2]);
      k_sa_fused<<<B * 512, 128, sm, stream>>>(xyz, nullptr, l1x, idxb,
          512, N, k1[br], 0, 0, Hsm1[br][0], Hsm1[br][1], Hsm1[br][2], l1p, 320, c1off[br]);
    }
  }

  // ----- level 2 (npoint=128) -----
  k_fps<<<B, 256, 0, stream>>>(l1x, 512, 128, fidx2);
  { int tot = B * 128;
    k_gather<<<(tot + 255) / 256, 256, 0, stream>>>(l1x, fidx2, 512, 128, tot, l2x); }
  {
    float r2v[3] = {0.2f, 0.4f, 0.8f};
    int   k2[3] = {32, 64, 128};
    int   c2off[3] = {0, 128, 384};
    for (int br = 0; br < 3; ++br) {
      k_ballquery<<<B * 128, 32, 0, stream>>>(l1x, l2x, 512, 128, k2[br], r2v[br] * r2v[br], idxb);
      size_t sm = smemB(k2[br], Hsm2[br][0], Hsm2[br][1], Hsm2[br][2]);
      k_sa_fused<<<B * 128, 128, sm, stream>>>(l1x, l1p, l2x, idxb,
          128, 512, k2[br], 320, 0, Hsm2[br][0], Hsm2[br][1], Hsm2[br][2], l2p, 640, c2off[br]);
    }
  }

  // ----- global SA stage (one group of 128 points per batch) -----
  {
    size_t sm = smemB(128, Hsa[0], Hsa[1], Hsa[2]);
    k_sa_fused<<<B, 128, sm, stream>>>(l2x, l2p, nullptr, nullptr,
        1, 128, 128, 640, 1, Hsa[0], Hsa[1], Hsa[2], feat, 1024, 0);
  }

  // ----- classifier head -----
  { int tot = B * 512;
    k_linear_bn_relu<<<(tot + 127) / 128, 128, 0, stream>>>(feat, lin1, B, 1024, h1); }
  { int tot = B * 256;
    k_linear_bn_relu<<<(tot + 127) / 128, 128, 0, stream>>>(h1, lin2, B, 512, h2); }
  k_head<<<B, 64, 0, stream>>>(h2, lin3_w, lin3_b, (float*)d_out);
  { int tot = B * 1024;
    k_copy<<<(tot + 255) / 256, 256, 0, stream>>>(feat, (float*)d_out + B * 40, tot); }
}
